// VirtualNodeUpdate_25975962206499
// MI455X (gfx1250) — matrix operand order repaired and checked by reading, not compiled, not run on hardware
//
#include <hip/hip_runtime.h>
#include <hip/hip_bf16.h>

#define Nn 100000
#define Hh 512
#define Bb 128

typedef _Float16 v16h __attribute__((ext_vector_type(16)));
typedef _Float16 v8h  __attribute__((ext_vector_type(8)));
typedef float    v8f  __attribute__((ext_vector_type(8)));

__device__ __forceinline__ v16h cat8(v8h lo, v8h hi) {
    return __builtin_shufflevector(lo, hi, 0,1,2,3,4,5,6,7,8,9,10,11,12,13,14,15);
}

__device__ __forceinline__ float silu(float v) {
    return v / (1.0f + __expf(-v));
}

// ---------------- zero workspace region (node_sum + counts) ----------------
__global__ void zero_kernel(float* p, int n) {
    int i = blockIdx.x * 256 + threadIdx.x;
    if (i < n) p[i] = 0.0f;
}

// ---------------- convert W3 -> W3t (f16, [512,1024]), W4 -> W4t ([512,512]) ----
__global__ void wconv_kernel(const float* __restrict__ W3, const float* __restrict__ W4,
                             _Float16* __restrict__ W3t, _Float16* __restrict__ W4t) {
    int idx = blockIdx.x * 256 + threadIdx.x;       // 0 .. 786431
    if (idx < 512 * 1024) {
        int j = idx >> 10, k = idx & 1023;          // W3t[j,k] = W3[k,j]
        W3t[idx] = (_Float16)W3[k * 512 + j];
    } else {
        int i = idx - 512 * 1024;                   // 0 .. 262143
        int j = i >> 9, k = i & 511;                // W4t[j,k] = W4[k,j]
        W4t[i] = (_Float16)W4[k * 512 + j];
    }
}

// ---------------- segment sum over sorted batch -------------------------------
#define RPB 128
__global__ void segsum_kernel(const float* __restrict__ x, const int* __restrict__ batch,
                              float* __restrict__ seg_sum, float* __restrict__ counts, int n) {
    int col = threadIdx.x;                  // 0..255, handles col and col+256
    int start = blockIdx.x * RPB;
    int end = start + RPB; if (end > n) end = n;
    float a0 = 0.0f, a1 = 0.0f, c = 0.0f;
    int cur = batch[start];
    for (int i = start; i < end; ++i) {
        int b = batch[i];
        if (b != cur) {
            atomicAdd(&seg_sum[cur * Hh + col], a0);
            atomicAdd(&seg_sum[cur * Hh + col + 256], a1);
            if (col == 0) atomicAdd(&counts[cur], c);
            a0 = a1 = c = 0.0f; cur = b;
        }
        a0 += x[(size_t)i * Hh + col];
        a1 += x[(size_t)i * Hh + col + 256];
        c += 1.0f;
    }
    atomicAdd(&seg_sum[cur * Hh + col], a0);
    atomicAdd(&seg_sum[cur * Hh + col + 256], a1);
    if (col == 0) atomicAdd(&counts[cur], c);
}

// ---------------- vn MLP layer 1: h1 = silu([node_mean, vn] @ W1 + b1) --------
__global__ void vn_mlp1_kernel(const float* __restrict__ seg_sum, const float* __restrict__ counts,
                               const float* __restrict__ vn, const float* __restrict__ W1,
                               const float* __restrict__ b1, float* __restrict__ h1) {
    __shared__ float a[2 * Hh];
    int b = blockIdx.x, t = threadIdx.x;
    float invc = 1.0f / fmaxf(counts[b], 1.0f);
    for (int i = t; i < Hh; i += 256) {
        a[i]      = seg_sum[b * Hh + i] * invc;
        a[Hh + i] = vn[b * Hh + i];
    }
    __syncthreads();
    for (int j = t; j < Hh; j += 256) {
        float s = b1[j];
        for (int k = 0; k < 2 * Hh; ++k) s += a[k] * W1[k * Hh + j];
        h1[b * Hh + j] = silu(s);
    }
}

// ---------------- vn MLP layer 2 + residual + LayerNorm -> vn_new -------------
__global__ void vn_mlp2_kernel(const float* __restrict__ h1, const float* __restrict__ vn,
                               const float* __restrict__ W2, const float* __restrict__ b2,
                               const float* __restrict__ g, const float* __restrict__ beta,
                               float* __restrict__ vn_new) {
    __shared__ float h[Hh];
    __shared__ float y[Hh];
    __shared__ float r1[256], r2[256];
    int b = blockIdx.x, t = threadIdx.x;
    for (int i = t; i < Hh; i += 256) h[i] = h1[b * Hh + i];
    __syncthreads();
    float s1 = 0.0f, s2 = 0.0f;
    for (int j = t; j < Hh; j += 256) {
        float s = b2[j];
        for (int k = 0; k < Hh; ++k) s += h[k] * W2[k * Hh + j];
        float yv = vn[b * Hh + j] + s;
        y[j] = yv; s1 += yv; s2 += yv * yv;
    }
    r1[t] = s1; r2[t] = s2;
    __syncthreads();
    for (int s = 128; s > 0; s >>= 1) {
        if (t < s) { r1[t] += r1[t + s]; r2[t] += r2[t + s]; }
        __syncthreads();
    }
    float mu = r1[0] * (1.0f / Hh);
    float var = r2[0] * (1.0f / Hh) - mu * mu;
    float rs = rsqrtf(var + 1e-5f);
    for (int j = t; j < Hh; j += 256)
        vn_new[b * Hh + j] = (y[j] - mu) * rs * g[j] + beta[j];
}

// ---------------- fused node update: WMMA GEMM1 -> SiLU -> WMMA GEMM2 -> LN ---
// 64 rows per workgroup; 8 waves = 4 row-groups x 2 col-groups;
// each wave owns 16 n-tiles (256 columns); one A-fragment feeds 16 WMMAs per K-step.
// LDS: Abuf 128 KB (f16 concat tile), aliased after GEMM1 by Hbuf (64 KB, f16 hidden)
// and Ybuf (64 KB, f32 residual for one 32-row half). Epilogue runs in two halves.
__global__ __launch_bounds__(256)
void node_update_kernel(const float* __restrict__ x, const int* __restrict__ batch,
                        const float* __restrict__ vn_new,
                        const _Float16* __restrict__ W3t, const _Float16* __restrict__ W4t,
                        const float* __restrict__ b3, const float* __restrict__ b4,
                        const float* __restrict__ g, const float* __restrict__ beta,
                        float* __restrict__ xout) {
    __shared__ _Float16 Abuf[64 * 1024];            // 128 KB
    __shared__ float red1[256], red2[256];
    __shared__ float stats[32 * 2];
    _Float16* Hbuf = Abuf;                          // 64*512 f16 = 64 KB (first half)
    float* Ybuf = (float*)(Abuf + 32768);           // 32*512 f32 = 64 KB (second half)

    const int t = threadIdx.x;
    const int wave = t >> 5;
    const int lane = t & 31;
    const int rg   = wave >> 1;             // row-group 0..3: rows rg*16 .. rg*16+15
    const int cg   = wave & 1;              // col-group 0..1: cols cg*256 .. cg*256+255
    const int base = blockIdx.x * 64;       // node row base (grid 1563; last WG partial)

    // --- stage A = [x | vn_new[batch]] as f16 into LDS (row-clamped) ----------
    #pragma unroll 4
    for (int i = 0; i < 256; ++i) {
        int e = t + i * 256;                // 0..65535
        int r = e >> 10, c = e & 1023;
        int node = base + r; if (node >= Nn) node = Nn - 1;
        float v = (c < Hh) ? x[(size_t)node * Hh + c]
                           : vn_new[(size_t)batch[node] * Hh + (c - Hh)];
        Abuf[e] = (_Float16)v;
    }
    __syncthreads();

    const int ma    = rg * 16 + (lane & 15); // A-fragment row
    const int kb    = (lane >> 4) * 8;       // A-fragment K sub-base
    const int koffB = (lane >> 4) * 8;       // B-fragment K sub-base
    const int ncol0 = cg * 256 + (lane & 15);
    const int mrow0 = rg * 16 + ((lane >> 4) << 3);  // D row base for this lane

    // --- GEMM1: h2 = silu(A[64x1024] @ W3 + b3), K-loop of 32 -----------------
    v8f acc[16];
    #pragma unroll
    for (int nt = 0; nt < 16; ++nt) acc[nt] = v8f{};
    for (int kk = 0; kk < 1024; kk += 32) {
        const _Float16* ap = &Abuf[ma * 1024 + kk + kb];
        v16h afrag = cat8(*(const v8h*)ap, *(const v8h*)(ap + 16));
        #pragma unroll
        for (int nt = 0; nt < 16; ++nt) {
            const _Float16* bp = W3t + (size_t)(ncol0 + nt * 16) * 1024 + kk + koffB;
            v16h bfrag = cat8(*(const v8h*)bp, *(const v8h*)(bp + 16));
            acc[nt] = __builtin_amdgcn_wmma_f32_16x16x32_f16(
                false, afrag, false, bfrag, (short)0, acc[nt], false, false);
        }
    }
    __syncthreads();                        // all waves done reading Abuf
    #pragma unroll
    for (int nt = 0; nt < 16; ++nt) {
        int n = ncol0 + nt * 16;
        float bb = b3[n];
        #pragma unroll
        for (int r = 0; r < 8; ++r) {
            Hbuf[(mrow0 + r) * 512 + n] = (_Float16)silu(acc[nt][r] + bb);
        }
    }
    __syncthreads();

    // --- GEMM2: node_upd = Hbuf[64x512] @ W4, K-loop of 16 --------------------
    v8f acc2[16];
    #pragma unroll
    for (int nt = 0; nt < 16; ++nt) acc2[nt] = v8f{};
    for (int kk = 0; kk < 512; kk += 32) {
        const _Float16* ap = &Hbuf[ma * 512 + kk + kb];
        v16h afrag = cat8(*(const v8h*)ap, *(const v8h*)(ap + 16));
        #pragma unroll
        for (int nt = 0; nt < 16; ++nt) {
            const _Float16* bp = W4t + (size_t)(ncol0 + nt * 16) * 512 + kk + koffB;
            v16h bfrag = cat8(*(const v8h*)bp, *(const v8h*)(bp + 16));
            acc2[nt] = __builtin_amdgcn_wmma_f32_16x16x32_f16(
                false, afrag, false, bfrag, (short)0, acc2[nt], false, false);
        }
    }

    // --- epilogue in two 32-row halves: residual + LayerNorm ------------------
    for (int half = 0; half < 2; ++half) {
        int hbase = half * 32;
        __syncthreads();                    // Ybuf free (Hbuf reads done / prev half done)
        // preload residual x (coalesced, row-clamped; out-of-range rows are dummies)
        #pragma unroll 4
        for (int i = 0; i < 64; ++i) {
            int e = t + i * 256;            // 0..16383
            int r = e >> 9, c = e & 511;
            int node = base + hbase + r; if (node >= Nn) node = Nn - 1;
            Ybuf[e] = x[(size_t)node * Hh + c];
        }
        __syncthreads();
        if ((rg >> 1) == half) {            // rg 0,1 -> rows 0..31; rg 2,3 -> rows 32..63
            #pragma unroll
            for (int nt = 0; nt < 16; ++nt) {
                int n = ncol0 + nt * 16;
                float bb = b4[n];
                #pragma unroll
                for (int r = 0; r < 8; ++r) {
                    Ybuf[(mrow0 - hbase + r) * 512 + n] += acc2[nt][r] + bb;
                }
            }
        }
        __syncthreads();
        // LayerNorm over 32 rows (8 threads/row)
        {
            int r = t >> 3, sub = t & 7;
            float s1 = 0.0f, s2 = 0.0f;
            for (int c = sub * 64; c < sub * 64 + 64; ++c) {
                float v = Ybuf[r * 512 + c];
                s1 += v; s2 += v * v;
            }
            red1[t] = s1; red2[t] = s2;
        }
        __syncthreads();
        if (t < 32) {
            float S1 = 0.0f, S2 = 0.0f;
            for (int i = 0; i < 8; ++i) { S1 += red1[t * 8 + i]; S2 += red2[t * 8 + i]; }
            float mu = S1 * (1.0f / Hh);
            float var = S2 * (1.0f / Hh) - mu * mu;
            stats[t * 2] = mu;
            stats[t * 2 + 1] = rsqrtf(var + 1e-5f);
        }
        __syncthreads();
        #pragma unroll 4
        for (int i = 0; i < 64; ++i) {
            int e = t + i * 256;
            int r = e >> 9, c = e & 511;
            int node = base + hbase + r;
            if (node < Nn) {
                float mu = stats[r * 2], rs = stats[r * 2 + 1];
                xout[(size_t)node * Hh + c] = (Ybuf[e] - mu) * rs * g[c] + beta[c];
            }
        }
    }
}

// ------------------------------------------------------------------------------
extern "C" void kernel_launch(void* const* d_in, const int* in_sizes, int n_in,
                              void* d_out, int out_size, void* d_ws, size_t ws_size,
                              hipStream_t stream) {
    const float* x       = (const float*)d_in[0];
    const float* vn      = (const float*)d_in[1];
    const float* W1      = (const float*)d_in[2];
    const float* b1      = (const float*)d_in[3];
    const float* W2      = (const float*)d_in[4];
    const float* b2      = (const float*)d_in[5];
    const float* g_vn    = (const float*)d_in[6];
    const float* beta_vn = (const float*)d_in[7];
    const float* W3      = (const float*)d_in[8];
    const float* b3      = (const float*)d_in[9];
    const float* W4      = (const float*)d_in[10];
    const float* b4      = (const float*)d_in[11];
    const float* g_nd    = (const float*)d_in[12];
    const float* beta_nd = (const float*)d_in[13];
    const int*   batch   = (const int*)d_in[14];

    float* xout   = (float*)d_out;
    float* vn_new = xout + (size_t)Nn * Hh;     // second tuple output, also read by node kernel

    // workspace layout (floats): node_sum[65536] | counts[128](pad 256) | h1[65536] | f16 W3t | f16 W4t
    float* ws       = (float*)d_ws;
    float* node_sum = ws;
    float* counts   = ws + 65536;
    float* h1       = ws + 65792;
    _Float16* W3t   = (_Float16*)(ws + 131328);
    _Float16* W4t   = W3t + 512 * 1024;

    zero_kernel<<<257, 256, 0, stream>>>(ws, 65664);                       // node_sum + counts
    wconv_kernel<<<3072, 256, 0, stream>>>(W3, W4, W3t, W4t);
    segsum_kernel<<<(Nn + RPB - 1) / RPB, 256, 0, stream>>>(x, batch, node_sum, counts, Nn);
    vn_mlp1_kernel<<<Bb, 256, 0, stream>>>(node_sum, counts, vn, W1, b1, h1);
    vn_mlp2_kernel<<<Bb, 256, 0, stream>>>(h1, vn, W2, b2, g_vn, beta_vn, vn_new);
    node_update_kernel<<<(Nn + 63) / 64, 256, 0, stream>>>(x, batch, vn_new, W3t, W4t,
                                                           b3, b4, g_nd, beta_nd, xout);
}
